// SkipLSTM_67585605369939
// MI455X (gfx1250) — matrix operand order repaired and checked by reading, not compiled, not run on hardware
//
#include <hip/hip_runtime.h>
#include <hip/hip_bf16.h>
#include <math.h>

// Problem dims (match reference)
#define B_   64
#define S_   256
#define IN_  3
#define H_   768
#define O_   121
#define L_   3
#define NW_  10
#define U_   80
#define C_   57
#define G4H  (4*H_)       // 3072 gate width
#define K0   (2*H_ + 64)  // 1600: [inp | h | w1pad] for layer 0
#define K12  (2*H_)       // 1536: [inp | h] for layers 1..2
#define CPAD 64
#define OPAD 128

// LDS B-panel row pitch in bf16 elements (32 data + 8 pad -> 80B rows: the
// 16 lanes of a half-wave hit disjoint 4-word bank groups on ds_load_b128).
#define BPITCH 40

typedef __bf16 bf16_t;
typedef __attribute__((ext_vector_type(8)))  __bf16 bf16x8;
typedef __attribute__((ext_vector_type(16))) __bf16 bf16x16;
typedef __attribute__((ext_vector_type(8)))  float  f32x8;

__device__ __forceinline__ bf16_t f2bf(float f) {
    unsigned u = __builtin_bit_cast(unsigned, f);
    u += 0x7FFFu + ((u >> 16) & 1u);           // round-to-nearest-even
    unsigned short h = (unsigned short)(u >> 16);
    return __builtin_bit_cast(bf16_t, h);
}
__device__ __forceinline__ float bf2f(bf16_t b) {
    unsigned short h = __builtin_bit_cast(unsigned short, b);
    return __builtin_bit_cast(float, ((unsigned)h) << 16);
}
__device__ __forceinline__ float sigmoidf(float x) { return 1.f / (1.f + expf(-x)); }
__device__ __forceinline__ bf16x16 cat16(bf16x8 lo, bf16x8 hi) {
    return __builtin_shufflevector(lo, hi, 0,1,2,3,4,5,6,7,8,9,10,11,12,13,14,15);
}

// ---------------------------------------------------------------------------
// bf16 WMMA GEMM: C[M,N] (f32 and/or bf16) = A[M,K] * B[N,K]^T (+C) (+bias[N])
// Block = 128 threads = 4 waves; block tile 64(M) x 64(N); wave w owns M-tile
// w; the 64x32 B k-panel is shared by all 4 waves via a double-buffered LDS
// stage. 2-deep software pipeline (unrolled x2): iteration i ds_loads the
// fragments of panel i+1 and global-loads panel i+2 while the WMMAs consume
// fragments loaded in iteration i-1 -> no LDS latency on the WMMA issue path.
// Requires M%64==0, N%64==0, K%64==0 (all call sites satisfy this); all
// branches are wave-uniform so EXEC is all-ones for WMMA and barriers.
// ---------------------------------------------------------------------------
__global__ __launch_bounds__(128)
void k_gemm_bf16(const bf16_t* __restrict__ A, int lda,
                 const bf16_t* __restrict__ Bm, int ldb,
                 float* __restrict__ C, int ldc,
                 bf16_t* __restrict__ Cbf, int ldcbf,
                 const float* __restrict__ bias,
                 int M, int N, int K, int accumulate)
{
    const int tid   = threadIdx.x;
    const int lane  = tid & 31;
    const int wave  = tid >> 5;                         // 0..3 -> M-tile
    const int mBase = blockIdx.y * 64 + wave * 16;
    const int nBase = blockIdx.x * 64;
    const int half  = lane >> 4;                        // 0|1
    const int l16   = lane & 15;

    __shared__ __align__(16) unsigned short sB[2][64 * BPITCH];

    f32x8 acc[4];
#pragma unroll
    for (int j = 0; j < 4; ++j) {
        if (accumulate && C) {
#pragma unroll
            for (int r = 0; r < 8; ++r) {
                int m = mBase + r + half * 8;
                int n = nBase + j * 16 + l16;
                acc[j][r] = C[(long)m * ldc + n];
            }
        } else {
#pragma unroll
            for (int r = 0; r < 8; ++r) acc[j][r] = 0.f;
        }
    }

    // Cooperative B staging: thread -> (row = tid/2, half-row = tid%2),
    // 16 bf16 (2 x b128) per thread per k-tile covers the 64x32 panel.
    const int ldRow  = tid >> 1;
    const int ldHalf = tid & 1;
    const bf16_t* bSrc = Bm + (long)(nBase + ldRow) * ldb + ldHalf * 16;
    unsigned short* sDst0 = &sB[0][ldRow * BPITCH + ldHalf * 16];
    unsigned short* sDst1 = &sB[1][ldRow * BPITCH + ldHalf * 16];

    const bf16_t* aRow = A + (long)(mBase + l16) * lda;

    // A 16x32 fragment (contiguous K, matching B): lanes 0-15 K {k..k+15},
    // lanes 16-31 K {k+16..k+31}.
    auto loadA = [&](int k) -> bf16x16 {
        const bf16_t* ap = aRow + k + half * 16;
        bf16x8 a0 = *reinterpret_cast<const bf16x8*>(ap);
        bf16x8 a1 = *reinterpret_cast<const bf16x8*>(ap + 8);
        return cat16(a0, a1);
    };
    // B 32x16 fragments from LDS: lane = column n, K contiguous per half-wave.
    auto loadB = [&](int buf, bf16x16* dst) {
#pragma unroll
        for (int j = 0; j < 4; ++j) {
            const unsigned short* lp = &sB[buf][(j * 16 + l16) * BPITCH + half * 16];
            bf16x8 b0 = *reinterpret_cast<const bf16x8*>(lp);
            bf16x8 b1 = *reinterpret_cast<const bf16x8*>(lp + 8);
            dst[j] = cat16(b0, b1);
        }
    };

    // ---- pipeline prologue: panel0 -> buf0, panel1 -> buf1 (K >= 64 always)
    bf16x8 gn0, gn1;
    {
        bf16x8 p0 = *reinterpret_cast<const bf16x8*>(bSrc);
        bf16x8 p1 = *reinterpret_cast<const bf16x8*>(bSrc + 8);
        *reinterpret_cast<bf16x8*>(sDst0)     = p0;
        *reinterpret_cast<bf16x8*>(sDst0 + 8) = p1;
        gn0 = *reinterpret_cast<const bf16x8*>(bSrc + 32);
        gn1 = *reinterpret_cast<const bf16x8*>(bSrc + 40);
    }
    __syncthreads();                                   // buf0 ready

    bf16x16 fA0, fA1, fB0[4], fB1[4];
    fA0 = loadA(0);
    loadB(0, fB0);
    *reinterpret_cast<bf16x8*>(sDst1)     = gn0;       // panel1 -> buf1
    *reinterpret_cast<bf16x8*>(sDst1 + 8) = gn1;
    __syncthreads();                                   // buf1 ready

    for (int k = 0; k < K; k += 64) {
        // even tile (panel i, frags already in fA0/fB0)
        if (k + 64 < K) {                              // global-load panel i+2
            const bf16_t* g = bSrc + (k + 64);
            __builtin_prefetch(g + 64, 0, 0);
            gn0 = *reinterpret_cast<const bf16x8*>(g);
            gn1 = *reinterpret_cast<const bf16x8*>(g + 8);
        }
        fA1 = loadA(k + 32);                           // frags of panel i+1
        loadB(1, fB1);
#pragma unroll
        for (int j = 0; j < 4; ++j)
            acc[j] = __builtin_amdgcn_wmma_f32_16x16x32_bf16(
                false, fA0, false, fB0[j], (short)0, acc[j], false, false);
        if (k + 64 < K) {                              // panel i+2 -> buf0
            *reinterpret_cast<bf16x8*>(sDst0)     = gn0;
            *reinterpret_cast<bf16x8*>(sDst0 + 8) = gn1;
            __syncthreads();
        }

        // odd tile (panel i+1, frags in fA1/fB1)
        if (k + 96 < K) {                              // global-load panel i+3
            const bf16_t* g = bSrc + (k + 96);
            gn0 = *reinterpret_cast<const bf16x8*>(g);
            gn1 = *reinterpret_cast<const bf16x8*>(g + 8);
        }
        if (k + 64 < K) {                              // frags of panel i+2
            fA0 = loadA(k + 64);
            loadB(0, fB0);
        }
#pragma unroll
        for (int j = 0; j < 4; ++j)
            acc[j] = __builtin_amdgcn_wmma_f32_16x16x32_bf16(
                false, fA1, false, fB1[j], (short)0, acc[j], false, false);
        if (k + 96 < K) {                              // panel i+3 -> buf1
            *reinterpret_cast<bf16x8*>(sDst1)     = gn0;
            *reinterpret_cast<bf16x8*>(sDst1 + 8) = gn1;
            __syncthreads();
        }
    }

#pragma unroll
    for (int j = 0; j < 4; ++j) {
        int n = nBase + j * 16 + l16;
        float bv = bias ? bias[n] : 0.f;
#pragma unroll
        for (int r = 0; r < 8; ++r) {
            int m = mBase + r + half * 8;
            float v = acc[j][r] + bv;
            if (C)   C  [(long)m * ldc   + n] = v;
            if (Cbf) Cbf[(long)m * ldcbf + n] = f2bf(v);
        }
    }
}

// ---------------------------------------------------------------------------
// Weight packing / small kernels
// ---------------------------------------------------------------------------
__global__ void k_pack_wcat(const float* __restrict__ Wih,
                            const float* __restrict__ Whh,
                            bf16_t* w0, bf16_t* w1, bf16_t* w2)
{
    long idx = blockIdx.x * (long)blockDim.x + threadIdx.x;
    if (idx >= 3L * G4H * K12) return;
    int  k = (int)(idx % K12);
    long ln = idx / K12;
    int  n = (int)(ln % G4H);
    int  l = (int)(ln / G4H);
    float v = (k < H_) ? Wih[((long)l * G4H + n) * H_ + k]
                       : Whh[((long)l * G4H + n) * H_ + (k - H_)];
    bf16_t* dst = (l == 0) ? w0 : ((l == 1) ? w1 : w2);
    int ld = (l == 0) ? K0 : K12;
    dst[(long)n * ld + k] = f2bf(v);
}

// WwhT[c,h] = W_wh[h,c] (c padded to 64), and Wwh_bf[h,c] (padded)
__global__ void k_pack_wwh(const float* __restrict__ Wwh, bf16_t* wT, bf16_t* wN)
{
    int idx = blockIdx.x * blockDim.x + threadIdx.x;
    if (idx >= H_ * CPAD) return;
    int c = idx % CPAD, h = idx / CPAD;
    float v = (c < C_) ? Wwh[h * C_ + c] : 0.f;
    bf16_t b = f2bf(v);
    wN[h * CPAD + c] = b;
    wT[(long)c * H_ + h] = b;
}

__global__ void k_pack_wfin(const float* __restrict__ Wfin, bf16_t* dst)
{
    long idx = blockIdx.x * (long)blockDim.x + threadIdx.x;
    if (idx >= 3L * OPAD * H_) return;
    int  h = (int)(idx % H_);
    long lo = idx / H_;
    int  o = (int)(lo % OPAD);
    int  l = (int)(lo / OPAD);
    float v = (o < O_) ? Wfin[((long)l * O_ + o) * H_ + h] : 0.f;
    dst[idx] = f2bf(v);
}

__global__ void k_bias_sum(const float* bih, const float* bhh, float* bsum)
{
    int idx = blockIdx.x * blockDim.x + threadIdx.x;
    if (idx >= 3 * G4H) return;
    bsum[idx] = bih[idx] + bhh[idx];
}

// xp[t,b,h] = x[b,t,:3] . W_in[h,:3] + b_in[h]; also seeds Abuf0 input columns
__global__ void k_input_proj(const float* __restrict__ x,
                             const float* __restrict__ Win,
                             const float* __restrict__ bin,
                             bf16_t* __restrict__ xp_bf,
                             bf16_t* __restrict__ abuf0)
{
    long idx = blockIdx.x * (long)blockDim.x + threadIdx.x;
    if (idx >= (long)S_ * B_ * H_) return;
    int  j = (int)(idx % H_);
    long tb = idx / H_;
    int  b = (int)(tb % B_);
    int  t = (int)(tb / B_);
    const float* xr = x + ((long)b * S_ + t) * IN_;
    const float* w  = Win + j * IN_;
    float v = bin[j] + xr[0]*w[0] + xr[1]*w[1] + xr[2]*w[2];
    bf16_t bf = f2bf(v);
    xp_bf[tb * H_ + j] = bf;
    abuf0[tb * K0 + j] = bf;
}

__global__ void k_zero_bf2d(bf16_t* p, int rows, int cols, int ld)
{
    int idx = blockIdx.x * blockDim.x + threadIdx.x;
    if (idx >= rows * cols) return;
    int r = idx / cols, c = idx % cols;
    p[(long)r * ld + c] = f2bf(0.f);
}

__global__ void k_zero_f32(float* p, long n)
{
    long idx = blockIdx.x * (long)blockDim.x + threadIdx.x;
    if (idx < n) p[idx] = 0.f;
}

// LSTM gate nonlinearity + state update; writes h as bf16 into the next step's
// A-panel recurrent columns and the per-layer h sequence (for output GEMM).
__global__ void k_lstm_gates(const float* __restrict__ G,  // [64,3072]
                             float* __restrict__ c,        // [64,768]
                             float* __restrict__ hf,       // [64,768]
                             bf16_t* __restrict__ hseq_bf, // + t*B*H
                             bf16_t* __restrict__ abuf_h,  // next-step h cols or null
                             int aK)
{
    int idx = blockIdx.x * blockDim.x + threadIdx.x;
    if (idx >= B_ * H_) return;
    int b = idx / H_, j = idx % H_;
    const float* g = G + (long)b * G4H;
    float ig = sigmoidf(g[j]);
    float fg = sigmoidf(g[H_ + j]);
    float gg = tanhf   (g[2*H_ + j]);
    float og = sigmoidf(g[3*H_ + j]);
    float cc = fg * c[idx] + ig * gg;
    c[idx] = cc;
    float h = og * tanhf(cc);
    hf[idx] = h;
    bf16_t hb = f2bf(h);
    hseq_bf[idx] = hb;
    if (abuf_h) abuf_h[(long)b * aK + j] = hb;
}

// Gaussian attention window for layer 0. One block per batch row.
__global__ __launch_bounds__(256)
void k_window(const float* __restrict__ hf,     // [64,768]
              const float* __restrict__ Wwin,   // [30,768]
              const float* __restrict__ bwin,   // [30]
              const float* __restrict__ seqs,   // [64,80,57]
              bf16_t* __restrict__ abuf_w,      // Abuf0[t+1] col 1536 (or null)
              bf16_t* __restrict__ w1bf)        // + t*B*CPAD
{
    __shared__ float sh[H_];
    __shared__ float sproj[32];
    __shared__ float sphi[U_];
    int b = blockIdx.x;
    for (int i = threadIdx.x; i < H_; i += 256) sh[i] = hf[(long)b * H_ + i];
    __syncthreads();
    if (threadIdx.x < 3 * NW_) {
        float s = bwin[threadIdx.x];
        const float* w = Wwin + threadIdx.x * H_;
        for (int i = 0; i < H_; ++i) s += sh[i] * w[i];
        sproj[threadIdx.x] = s;
    }
    __syncthreads();
    if (threadIdx.x < U_) {
        float u = (float)threadIdx.x;
        float p = 0.f;
        for (int q = 0; q < NW_; ++q) {
            float alpha = sproj[q];
            float beta  = expf(sproj[NW_ + q]);
            float keta  = expf(sproj[2*NW_ + q]);
            float d = keta - u;
            p += alpha - beta * d * d;
        }
        sphi[threadIdx.x] = p;
    }
    __syncthreads();
    if (threadIdx.x < CPAD) {
        int c = threadIdx.x;
        float s = 0.f;
        if (c < C_)
            for (int u = 0; u < U_; ++u) s += sphi[u] * seqs[((long)b * U_ + u) * C_ + c];
        bf16_t v = f2bf(s);
        w1bf[b * CPAD + c] = v;
        if (abuf_w) abuf_w[(long)b * K0 + c] = v;
    }
}

// inp_l = xp + hseq_{l-1} + wseq  (all [t,b,h] bf16) -> A-panel input columns
__global__ void k_sum_inputs(const bf16_t* __restrict__ xp,
                             const bf16_t* __restrict__ hprev,
                             const bf16_t* __restrict__ wseq,
                             bf16_t* __restrict__ abuf, int aK)
{
    long idx = blockIdx.x * (long)blockDim.x + threadIdx.x;
    if (idx >= (long)S_ * B_ * H_) return;
    int  j = (int)(idx % H_);
    long tb = idx / H_;
    float v = bf2f(xp[idx]) + bf2f(hprev[idx]) + bf2f(wseq[idx]);
    abuf[tb * aK + j] = f2bf(v);
}

// out[b,s,o] = OutScr[(s*B+b)*128 + o] + bias
__global__ void k_out_final(const float* __restrict__ outscr,
                            const float* __restrict__ bias,
                            float* __restrict__ dout)
{
    long idx = blockIdx.x * (long)blockDim.x + threadIdx.x;
    if (idx >= (long)B_ * S_ * O_) return;
    int  o = (int)(idx % O_);
    long bs = idx / O_;
    int  t = (int)(bs % S_);
    int  b = (int)(bs / S_);
    dout[idx] = outscr[((long)t * B_ + b) * OPAD + o] + bias[0];
}

__global__ void k_copy_f32(const float* __restrict__ s, float* __restrict__ d, long n)
{
    long idx = blockIdx.x * (long)blockDim.x + threadIdx.x;
    if (idx < n) d[idx] = s[idx];
}

// ---------------------------------------------------------------------------
// Host orchestration
// ---------------------------------------------------------------------------
static inline void gemm(hipStream_t s, const void* A, int lda, const void* B, int ldb,
                        float* C, int ldc, void* Cbf, int ldcbf, const float* bias,
                        int M, int N, int K, int acc)
{
    dim3 grid(N / 64, M / 64);
    k_gemm_bf16<<<grid, 128, 0, s>>>((const bf16_t*)A, lda, (const bf16_t*)B, ldb,
                                     C, ldc, (bf16_t*)Cbf, ldcbf, bias, M, N, K, acc);
}
static inline dim3 g1(long n, int blk) { return dim3((unsigned)((n + blk - 1) / blk)); }

extern "C" void kernel_launch(void* const* d_in, const int* in_sizes, int n_in,
                              void* d_out, int out_size, void* d_ws, size_t ws_size,
                              hipStream_t stream)
{
    const float* x    = (const float*)d_in[0];
    const float* seqs = (const float*)d_in[1];
    const float* Win  = (const float*)d_in[2];
    const float* bin  = (const float*)d_in[3];
    const float* Wih  = (const float*)d_in[4];
    const float* Whh  = (const float*)d_in[5];
    const float* bih  = (const float*)d_in[6];
    const float* bhh  = (const float*)d_in[7];
    const float* Wfin = (const float*)d_in[8];
    const float* Wwin = (const float*)d_in[9];
    const float* bwin = (const float*)d_in[10];
    const float* Wwh  = (const float*)d_in[11];
    const float* bias = (const float*)d_in[12];
    float* out = (float*)d_out;

    // ---- workspace carve-out (256B aligned) ----
    char* p = (char*)d_ws;
    auto take = [&](size_t bytes) -> char* {
        char* r = p; p += (bytes + 255) & ~(size_t)255; return r;
    };
    bf16_t* Wcat0  = (bf16_t*)take(2L * G4H * K0);          // [3072,1600]
    bf16_t* Wcat1  = (bf16_t*)take(2L * G4H * K12);         // [3072,1536]
    bf16_t* Wcat2  = (bf16_t*)take(2L * G4H * K12);
    bf16_t* WwhT   = (bf16_t*)take(2L * CPAD * H_);         // [64,768]
    bf16_t* WwhN   = (bf16_t*)take(2L * H_ * CPAD);         // [768,64]
    bf16_t* WfinBf = (bf16_t*)take(2L * 3 * OPAD * H_);     // [3,128,768]
    float*  bsum   = (float*) take(4L * 3 * G4H);
    bf16_t* xpBf   = (bf16_t*)take(2L * S_ * B_ * H_);
    bf16_t* Abuf0  = (bf16_t*)take(2L * S_ * B_ * K0);
    bf16_t* Abuf1  = (bf16_t*)take(2L * S_ * B_ * K12);
    bf16_t* Abuf2  = (bf16_t*)take(2L * S_ * B_ * K12);
    bf16_t* hseq0  = (bf16_t*)take(2L * S_ * B_ * H_);
    bf16_t* hseq1  = (bf16_t*)take(2L * S_ * B_ * H_);
    bf16_t* hseq2  = (bf16_t*)take(2L * S_ * B_ * H_);
    bf16_t* wseqBf = (bf16_t*)take(2L * S_ * B_ * H_);
    bf16_t* W1bf   = (bf16_t*)take(2L * S_ * B_ * CPAD);
    float*  Gcur   = (float*) take(4L * B_ * G4H);
    float*  cCur   = (float*) take(4L * L_ * B_ * H_);      // [3,64,768]
    float*  hCur   = (float*) take(4L * L_ * B_ * H_);
    float*  OutScr = (float*) take(4L * S_ * B_ * OPAD);
    (void)ws_size; (void)in_sizes; (void)n_in; (void)out_size;

    // ---- weight packing ----
    k_pack_wcat<<<g1(3L*G4H*K12, 256), 256, 0, stream>>>(Wih, Whh, Wcat0, Wcat1, Wcat2);
    k_pack_wwh <<<g1(H_*CPAD, 256), 256, 0, stream>>>(Wwh, WwhT, WwhN);
    k_pack_wfin<<<g1(3L*OPAD*H_, 256), 256, 0, stream>>>(Wfin, WfinBf);
    k_bias_sum <<<g1(3*G4H, 256), 256, 0, stream>>>(bih, bhh, bsum);

    // Wc[3072,64] = Wih0 @ W_wh  (folded window->gate matrix), written straight
    // into A-panel K columns [1536,1600) of Wcat0. A aliases Wcat0 cols [0,768).
    gemm(stream, Wcat0, K0, WwhT, H_, nullptr, 0, Wcat0 + K12, K0, nullptr,
         G4H, CPAD, H_, 0);

    // ---- input projection + zero-init of step-0 recurrent/window columns ----
    k_input_proj<<<g1((long)S_*B_*H_, 256), 256, 0, stream>>>(x, Win, bin, xpBf, Abuf0);
    k_zero_bf2d<<<g1(B_*(K0 - H_), 256), 256, 0, stream>>>(Abuf0 + H_, B_, K0 - H_, K0);
    k_zero_bf2d<<<g1(B_*H_, 256), 256, 0, stream>>>(Abuf1 + H_, B_, H_, K12);
    k_zero_bf2d<<<g1(B_*H_, 256), 256, 0, stream>>>(Abuf2 + H_, B_, H_, K12);
    k_zero_f32<<<g1(2L*L_*B_*H_, 256), 256, 0, stream>>>(cCur, 2L*L_*B_*H_); // c & h

    // ---- layer 0 scan: fused [inp|h|w1] @ [Wih;Whh;Wc]^T per step ----
    for (int t = 0; t < S_; ++t) {
        gemm(stream, Abuf0 + (long)t*B_*K0, K0, Wcat0, K0,
             Gcur, G4H, nullptr, 0, bsum, B_, G4H, K0, 0);
        bf16_t* abufH = (t + 1 < S_) ? Abuf0 + (long)(t+1)*B_*K0 + H_  : nullptr;
        bf16_t* abufW = (t + 1 < S_) ? Abuf0 + (long)(t+1)*B_*K0 + K12 : nullptr;
        k_lstm_gates<<<g1(B_*H_, 256), 256, 0, stream>>>(
            Gcur, cCur, hCur, hseq0 + (long)t*B_*H_, abufH, K0);
        k_window<<<B_, 256, 0, stream>>>(hCur, Wwin, bwin, seqs,
                                         abufW, W1bf + (long)t*B_*CPAD);
    }

    // wseq[t,b,:] = w1[t,b,:] @ W_wh^T  -> bf16, batched over all steps
    gemm(stream, W1bf, CPAD, WwhN, CPAD, nullptr, 0, wseqBf, H_, nullptr,
         S_*B_, H_, CPAD, 0);

    // ---- layers 1..2 ----
    bf16_t* abufs[2] = { Abuf1, Abuf2 };
    bf16_t* hseqs[3] = { hseq0, hseq1, hseq2 };
    bf16_t* wcats[2] = { Wcat1, Wcat2 };
    for (int l = 1; l < L_; ++l) {
        bf16_t* Abuf = abufs[l-1];
        k_sum_inputs<<<g1((long)S_*B_*H_, 256), 256, 0, stream>>>(
            xpBf, hseqs[l-1], wseqBf, Abuf, K12);
        for (int t = 0; t < S_; ++t) {
            gemm(stream, Abuf + (long)t*B_*K12, K12, wcats[l-1], K12,
                 Gcur, G4H, nullptr, 0, bsum + l*G4H, B_, G4H, K12, 0);
            bf16_t* abufH = (t + 1 < S_) ? Abuf + (long)(t+1)*B_*K12 + H_ : nullptr;
            k_lstm_gates<<<g1(B_*H_, 256), 256, 0, stream>>>(
                Gcur, cCur + (long)l*B_*H_, hCur + (long)l*B_*H_,
                hseqs[l] + (long)t*B_*H_, abufH, K12);
        }
    }

    // ---- output projection: sum_l hseq_l @ Wfin_l^T (O padded to 128) ----
    for (int l = 0; l < L_; ++l)
        gemm(stream, hseqs[l], H_, WfinBf + (long)l*OPAD*H_, H_,
             OutScr, OPAD, nullptr, 0, nullptr, S_*B_, OPAD, H_, l > 0);

    k_out_final<<<g1((long)B_*S_*O_, 256), 256, 0, stream>>>(OutScr, bias, out);
    long stateN = (long)L_ * B_ * H_;
    k_copy_f32<<<g1(stateN, 256), 256, 0, stream>>>(hCur, out + (long)B_*S_*O_, stateN);
    k_copy_f32<<<g1(stateN, 256), 256, 0, stream>>>(cCur, out + (long)B_*S_*O_ + stateN, stateN);
}
